// Attention_82695300317344
// MI455X (gfx1250) — matrix operand order repaired and checked by reading, not compiled, not run on hardware
//
#include <hip/hip_runtime.h>
#include <hip/hip_bf16.h>

// ---- problem constants (B,H,S,D) = (4,8,2048,64), fp32 in/out ----
constexpr int Bz = 4, Hh = 8, Sq = 2048, Dd = 64;

typedef _Float16 v16h __attribute__((ext_vector_type(16)));
typedef _Float16 h8   __attribute__((ext_vector_type(8)));
typedef _Float16 h2   __attribute__((ext_vector_type(2)));
typedef float    v8f  __attribute__((ext_vector_type(8)));
typedef float    f4   __attribute__((ext_vector_type(4)));

// LDS layout (padded strides to dodge bank conflicts; 64 banks x 4B)
constexpr int SCW = 2048 + 16;   // f32 score row stride
constexpr int WSW = 2048 + 40;   // f16 weight row stride (4176 B rows, 16B aligned)
constexpr int SC_BYTES  = 16 * SCW * 4;   // 132096
constexpr int WG_BYTES  = 16 * WSW * 2;   //  66816
constexpr int RED_OFF   = SC_BYTES + WG_BYTES;
constexpr int SMEM_BYTES = RED_OFF + 2 * 16 * 8 * 4;  // 199936

// Phase-3 V staging overlay (reuses the dead score buffer):
// Vt[d][t_local] f16, 64 rows, TC=1024 cols, stride TVW=1032 halfs
// -> 64*1032*2 = 132096 bytes, exact fit in SC region.
constexpr int TC  = 1024;
constexpr int TVW = TC + 8;

static __device__ __forceinline__ v8f wmma_f16(v16h a, v16h b, v8f c) {
  return __builtin_amdgcn_wmma_f32_16x16x32_f16(
      false, a, false, b, (short)0, c, false, false);
}

// A-operand (16x32 f16) from row-major f32: K = 16*hh+i
static __device__ __forceinline__ v16h load_a_f32(const float* __restrict__ row,
                                                  int dbase, int hh) {
  const f4* p0 = (const f4*)(row + dbase + 16 * hh);
  const f4* p1 = (const f4*)(row + dbase + 16 * hh + 8);
  f4 x0 = p0[0], x1 = p0[1], y0 = p1[0], y1 = p1[1];
  v16h a;
#pragma unroll
  for (int i = 0; i < 4; ++i) {
    a[i] = (_Float16)x0[i];     a[4 + i]  = (_Float16)x1[i];
    a[8 + i] = (_Float16)y0[i]; a[12 + i] = (_Float16)y1[i];
  }
  return a;
}

// B-operand (32x16 f16) for X*Y^T: lanes 0-15 K=0..15, lanes 16-31 K=16..31
static __device__ __forceinline__ v16h load_b_f32(const float* __restrict__ row,
                                                  int dbase, int hh) {
  const f4* p = (const f4*)(row + dbase + 16 * hh);
  f4 x0 = p[0], x1 = p[1], x2 = p[2], x3 = p[3];
  v16h b;
#pragma unroll
  for (int i = 0; i < 4; ++i) {
    b[i] = (_Float16)x0[i];     b[4 + i]  = (_Float16)x1[i];
    b[8 + i] = (_Float16)x2[i]; b[12 + i] = (_Float16)x3[i];
  }
  return b;
}

// A-operand from contiguous f16 LDS row (same contiguous K order as load_a_f32)
static __device__ __forceinline__ v16h load_a_h16(const _Float16* __restrict__ wr,
                                                  int hh) {
  h8 lo = *(const h8*)(wr + 16 * hh);
  h8 hi = *(const h8*)(wr + 16 * hh + 8);
  v16h a;
#pragma unroll
  for (int i = 0; i < 8; ++i) { a[i] = lo[i]; a[8 + i] = hi[i]; }
  return a;
}

__global__ __launch_bounds__(128, 1) void attn_fwd_kernel(
    const float* __restrict__ Q, const float* __restrict__ K,
    const float* __restrict__ V, const int* __restrict__ padmask,
    float* __restrict__ outRes, float* __restrict__ outAttn) {
  extern __shared__ char smem[];
  float*    sc     = (float*)smem;                       // [16][SCW] f32 scores
  _Float16* wgt    = (_Float16*)(smem + SC_BYTES);       // [16][WSW] f16 weights
  float*    redmax = (float*)(smem + RED_OFF);           // [16][8]
  float*    redsum = redmax + 16 * 8;                    // [16][8]
  _Float16* Vt     = (_Float16*)smem;                    // phase-3 overlay of sc

  const int bh = blockIdx.x >> 7;
  const int qt = blockIdx.x & 127;
  const int b  = bh >> 3;
  const int q0 = qt * 16;

  const float* Qh = Q + (size_t)bh * Sq * Dd;
  const float* Kh = K + (size_t)bh * Sq * Dd;
  const float* Vh = V + (size_t)bh * Sq * Dd;
  const int*   pm = padmask + (size_t)b * Sq;
  float* attn = outAttn + (size_t)bh * Sq * Sq + (size_t)q0 * Sq;
  float* res  = outRes  + (size_t)bh * Sq * Dd + (size_t)q0 * Dd;

  const int tid  = threadIdx.x;
  const int lane = tid & 31, wave = tid >> 5;
  const int lrow = lane & 15, hh = lane >> 4;

  // ---------------- Phase 1: raw scores Q*K^T into LDS ----------------
  const float* qrow = Qh + (size_t)(q0 + lrow) * Dd;
  v16h aq0 = load_a_f32(qrow, 0, hh);
  v16h aq1 = load_a_f32(qrow, 32, hh);

  for (int kt = wave; kt < Sq / 16; kt += 4) {
    const float* krow = Kh + (size_t)(kt * 16 + lrow) * Dd;
    v16h bk0 = load_b_f32(krow, 0, hh);
    v16h bk1 = load_b_f32(krow, 32, hh);
    v8f c = {0.f, 0.f, 0.f, 0.f, 0.f, 0.f, 0.f, 0.f};
    c = wmma_f16(aq0, bk0, c);
    c = wmma_f16(aq1, bk1, c);
    float* dst = sc + (size_t)(8 * hh) * SCW + kt * 16 + lrow;  // c[v]=(v+8hh, lrow)
#pragma unroll
    for (int v = 0; v < 8; ++v) dst[(size_t)v * SCW] = c[v];
  }
  __syncthreads();

  // ---------------- Phase 2: scale, mask, softmax, emit weights -------
  const int r = tid >> 3, sub = tid & 7;     // 8 threads per query row
  const int srow = q0 + r;
  float* scr = sc + (size_t)r * SCW;

  float mx = -3.0e38f;
#pragma unroll 4
  for (int j = 0; j < Sq / 8; ++j) {
    int col = sub + 8 * j;
    float x = scr[col] * 0.125f;                 // 1/sqrt(64)
    int fut = (col > srow) ? 1 : 0;
    int pd  = pm[col];
    if (fut + pd == 1) x = -1.0e9f;              // overlap (==2) NOT masked
    scr[col] = x;
    mx = fmaxf(mx, x);
  }
  redmax[r * 8 + sub] = mx;
  __syncthreads();
  float rmax = redmax[r * 8];
#pragma unroll
  for (int i = 1; i < 8; ++i) rmax = fmaxf(rmax, redmax[r * 8 + i]);

  float sum = 0.f;
#pragma unroll 4
  for (int j = 0; j < Sq / 8; ++j) {
    int col = sub + 8 * j;
    float e = __expf(scr[col] - rmax);
    scr[col] = e;
    sum += e;
  }
  redsum[r * 8 + sub] = sum;
  __syncthreads();
  float tot = redsum[r * 8];
#pragma unroll
  for (int i = 1; i < 8; ++i) tot += redsum[r * 8 + i];
  const float inv = 1.0f / tot;

  _Float16* wrow = wgt + (size_t)r * WSW;
  float* arow = attn + (size_t)r * Sq;
#pragma unroll 4
  for (int j = 0; j < Sq / 8; ++j) {
    int col = sub + 8 * j;
    float w = scr[col] * inv;
    arow[col] = w;                   // 512 MB output: the bandwidth roofline
    wrow[col] = (_Float16)w;         // f16 copy for PV WMMA
  }

  // ---------------- Phase 3: result = W * V ---------------------------
  // V staged per 1024-row chunk into LDS as f16, TRANSPOSED (Vt[d][t]),
  // so the WMMA B-operand is two contiguous 16B ds loads per lane.
  const int d0   = wave * 16;
  const int cgrp = tid & 15;        // d-group: d = 4*cgrp + 0..3
  const int tpr  = tid >> 4;        // t-pair selector
  v8f acc = {0.f, 0.f, 0.f, 0.f, 0.f, 0.f, 0.f, 0.f};

  for (int ch = 0; ch < Sq / TC; ++ch) {
    const int t0 = ch * TC;
    __syncthreads();                 // previous users of the overlay done

    // stage: coalesced f32 row loads (16 lanes * float4 = full 256B row),
    // pack (t, t+1) half pairs -> one b32 store per d
    for (int j = 0; j < TC / 16; ++j) {
      int tl = 2 * tpr + 16 * j;     // even local t
      f4 x0 = *(const f4*)(Vh + (size_t)(t0 + tl) * Dd + 4 * cgrp);
      f4 x1 = *(const f4*)(Vh + (size_t)(t0 + tl + 1) * Dd + 4 * cgrp);
#pragma unroll
      for (int i = 0; i < 4; ++i) {
        h2 p; p[0] = (_Float16)x0[i]; p[1] = (_Float16)x1[i];
        *(h2*)(Vt + (size_t)(4 * cgrp + i) * TVW + tl) = p;
      }
    }
    __syncthreads();

    // compute: b[i] = V[t0 + kt*32 + 16*hh + i][d0 + lrow] = contiguous in Vt
    for (int kt = 0; kt < TC / 32; ++kt) {
      const _Float16* wr = wgt + (size_t)lrow * WSW + t0 + kt * 32;
      v16h aw = load_a_h16(wr, hh);
      const _Float16* vb = Vt + (size_t)(d0 + lrow) * TVW + kt * 32 + 16 * hh;
      h8 blo = *(const h8*)vb;
      h8 bhi = *(const h8*)(vb + 8);
      v16h bv;
#pragma unroll
      for (int i = 0; i < 8; ++i) { bv[i] = blo[i]; bv[8 + i] = bhi[i]; }
      acc = wmma_f16(aw, bv, acc);
    }
  }

  float* rp = res + (size_t)(8 * hh) * Dd + d0 + lrow;   // acc[v]=(v+8hh, lrow)
#pragma unroll
  for (int v = 0; v < 8; ++v) rp[(size_t)v * Dd] = acc[v];
}

extern "C" void kernel_launch(void* const* d_in, const int* in_sizes, int n_in,
                              void* d_out, int out_size, void* d_ws, size_t ws_size,
                              hipStream_t stream) {
  (void)in_sizes; (void)n_in; (void)out_size; (void)d_ws; (void)ws_size;
  const float* q  = (const float*)d_in[0];
  const float* k  = (const float*)d_in[1];
  const float* v  = (const float*)d_in[2];
  const int*   pm = (const int*)d_in[3];

  float* outRes  = (float*)d_out;                           // [B,H,S,D]
  float* outAttn = outRes + (size_t)Bz * Hh * Sq * Dd;      // [B,H,S,S]

  (void)hipFuncSetAttribute(reinterpret_cast<const void*>(attn_fwd_kernel),
                            hipFuncAttributeMaxDynamicSharedMemorySize,
                            SMEM_BYTES);

  dim3 grid(Bz * Hh * (Sq / 16));   // 4096 workgroups
  dim3 block(128);                  // 4 waves (wave32)
  attn_fwd_kernel<<<grid, block, SMEM_BYTES, stream>>>(q, k, v, pm, outRes, outAttn);
}
